// SparseAttention_87222195847292
// MI455X (gfx1250) — compile-verified, matrix operand order checked by reading, not run on hardware
//
#include <hip/hip_runtime.h>
#include <hip/hip_bf16.h>
#include <math.h>

// ---------------------------------------------------------------------------
// SparseAttention, restructured:
//   1) xsum[b,blk,:]  = sum of x rows per 256-row block        (HBM read 33.5MB)
//   2) Sv   = xsum @ W_v + 256*b_v          (WMMA f32, 32x1024x1024)
//   3) attn = c1*Sv + c2*sum_blk(Sv)        (softmax collapses to 2 constants)
//   4) Y    = attn @ W_out + b_out          (WMMA f32, 32x1024x1024)
//   5) out[b,i,:] = Y[b,i/256,:]                                (HBM write 33.5MB)
// Total ~134 MFLOP, ~75 MB traffic -> memory-bound at 23.3 TB/s.
// ---------------------------------------------------------------------------

typedef float    v2f  __attribute__((ext_vector_type(2)));
typedef float    v4f  __attribute__((ext_vector_type(4)));
typedef float    v8f  __attribute__((ext_vector_type(8)));
typedef _Float16 v16h __attribute__((ext_vector_type(16)));

#ifndef __has_builtin
#define __has_builtin(x) 0
#endif
#if __has_builtin(__builtin_amdgcn_wmma_f32_16x16x4_f32)
#define USE_WMMA_F32 1
#else
#define USE_WMMA_F32 0
#endif

#define D_MODEL 1024
#define NBLK_TOT 32   // B(2) * blocks-per-batch(16)
#define RSPLIT 8      // row-slices per block for the reduction pass

// ---- Pass 1a: partial block sums of x (streaming, nontemporal loads) ------
__global__ __launch_bounds__(256) void k_blocksum_partial(
    const float* __restrict__ x, float* __restrict__ partial) {
  const int wg = blockIdx.x;          // 0..255  = bb*8 + rs
  const int rs = wg & (RSPLIT - 1);
  const int bb = wg >> 3;             // b*16 + blk
  const v4f* src =
      (const v4f*)(x + ((size_t)bb * 256 + (size_t)rs * 32) * D_MODEL) +
      threadIdx.x;                    // 256 threads cover 1024 floats
  v4f acc = {0.f, 0.f, 0.f, 0.f};
#pragma unroll 8
  for (int r = 0; r < 32; ++r) {
    v4f v = __builtin_nontemporal_load(src + r * (D_MODEL / 4));
    acc += v;
  }
  ((v4f*)partial)[(size_t)wg * (D_MODEL / 4) + threadIdx.x] = acc;
}

// ---- Pass 1b: combine partials -> xsum[32,1024] ---------------------------
__global__ __launch_bounds__(256) void k_blocksum_combine(
    const float* __restrict__ partial, float* __restrict__ xsum) {
  const int i  = blockIdx.x * 256 + threadIdx.x;  // 0..8191 v4f slots
  const int bb = i >> 8;
  const int c4 = i & 255;
  v4f acc = {0.f, 0.f, 0.f, 0.f};
#pragma unroll
  for (int rs = 0; rs < RSPLIT; ++rs) {
    acc += ((const v4f*)partial)[((size_t)bb * RSPLIT + rs) * 256 + c4];
  }
  ((v4f*)xsum)[i] = acc;
}

// ---- WMMA GEMM: C[32,1024] = A[32,1024] @ B[:,coloff:coloff+1024] + s*bias -
// One wave per 16x16 output tile; 128 tiles = 16 WGs x 8 waves.
__global__ __launch_bounds__(256) void k_gemm_wmma(
    const float* __restrict__ A, const float* __restrict__ Bm, int ldb,
    int coloff, const float* __restrict__ bias, int bias_off, float bias_scale,
    float* __restrict__ Cout) {
#if defined(__HIP_DEVICE_COMPILE__)
  const int lane = threadIdx.x & 31;
  const int wave = threadIdx.x >> 5;
  const int tile = blockIdx.x * 8 + wave;  // 0..127
  const int mt = tile >> 6;                // 0..1   (M tile)
  const int nt = tile & 63;                // 0..63  (N tile)
  const int m  = lane & 15;
  const int h  = lane >> 4;
  const int col = nt * 16 + m;             // lane%16 indexes N for B and C/D
  v8f c = {0.f, 0.f, 0.f, 0.f, 0.f, 0.f, 0.f, 0.f};

#if USE_WMMA_F32
  // A 16x4 f32 fragment: lane(m, h): {A[m][k+2h], A[m][k+2h+1]}
  const float* arow = A + (size_t)(mt * 16 + m) * D_MODEL + 2 * h;
  const float* bcol = Bm + coloff + col;
#pragma unroll 4
  for (int k = 0; k < D_MODEL; k += 4) {
    v2f a = *(const v2f*)(arow + k);
    v2f b;
    b.x = bcol[(size_t)(k + 2 * h) * ldb];
    b.y = bcol[(size_t)(k + 2 * h + 1) * ldb];
    c = __builtin_amdgcn_wmma_f32_16x16x4_f32(false, a, false, b, (short)0, c,
                                              false, false);
  }
#else
  // Fallback: f16 inputs, f32 accumulate, K-step 32 (codegen-confirmed builtin)
  const float* arow = A + (size_t)(mt * 16 + m) * D_MODEL;
  const float* bcol = Bm + coloff + col;
  for (int k = 0; k < D_MODEL; k += 32) {
    v16h a, b;
#pragma unroll
    for (int j = 0; j < 8; ++j) {
      int kb = (j < 4) ? (2 * j + 8 * h) : (16 + 2 * (j - 4) + 8 * h);
      a[2 * j]     = (_Float16)arow[k + kb];
      a[2 * j + 1] = (_Float16)arow[k + kb + 1];
      b[2 * j]     = (_Float16)bcol[(size_t)(k + kb) * ldb];
      b[2 * j + 1] = (_Float16)bcol[(size_t)(k + kb + 1) * ldb];
    }
    c = __builtin_amdgcn_wmma_f32_16x16x32_f16(false, a, false, b, (short)0, c,
                                               false, false);
  }
#endif
  // C/D layout: D[v + 8*h][lane%16] = c[v]
  const float bz = bias_scale * bias[bias_off + col];
#pragma unroll
  for (int v = 0; v < 8; ++v) {
    const int row = mt * 16 + v + 8 * h;
    Cout[(size_t)row * D_MODEL + col] = c[v] + bz;
  }
#endif  // __HIP_DEVICE_COMPILE__
}

// ---- Collapsed softmax: attn[b,blk,:] = c1*Sv[b,blk,:] + c2*sum_blk Sv ----
__global__ __launch_bounds__(256) void k_attn(
    const float* __restrict__ Sv, float* __restrict__ attn, float c1, float c2) {
  const int i = blockIdx.x * 256 + threadIdx.x;  // 0..2047 = b*1024 + d
  const int b = i >> 10;
  const int d = i & 1023;
  const float* base = Sv + (size_t)b * 16 * D_MODEL + d;
  float v[16];
  float sall = 0.f;
#pragma unroll
  for (int blk = 0; blk < 16; ++blk) {
    v[blk] = base[(size_t)blk * D_MODEL];
    sall += v[blk];
  }
  float* ob = attn + (size_t)b * 16 * D_MODEL + d;
#pragma unroll
  for (int blk = 0; blk < 16; ++blk)
    ob[(size_t)blk * D_MODEL] = c1 * v[blk] + c2 * sall;
}

// ---- Broadcast Y[b,blk,:] to all 256 rows of the block (NT stores) --------
__global__ __launch_bounds__(256) void k_broadcast(
    const float* __restrict__ Y, float* __restrict__ out) {
  const int row = blockIdx.x;        // 0..8191 global row = b*4096 + n
  const int bb  = row >> 8;          // = b*16 + n/256
  v4f v = ((const v4f*)(Y + (size_t)bb * D_MODEL))[threadIdx.x];
  v4f* dst = (v4f*)(out + (size_t)row * D_MODEL) + threadIdx.x;
  __builtin_nontemporal_store(v, dst);
}

extern "C" void kernel_launch(void* const* d_in, const int* in_sizes, int n_in,
                              void* d_out, int out_size, void* d_ws,
                              size_t ws_size, hipStream_t stream) {
  const float* x     = (const float*)d_in[0];  // (2,4096,1024)
  const float* W_qkv = (const float*)d_in[1];  // (1024,3072)
  const float* b_qkv = (const float*)d_in[2];  // (3072,)
  const float* W_out = (const float*)d_in[3];  // (1024,1024)
  const float* b_out = (const float*)d_in[4];  // (1024,)
  float* out = (float*)d_out;                  // (2,4096,1024)

  float* ws      = (float*)d_ws;               // 1.5 MB used
  float* partial = ws;                         // 32*8*1024
  float* xsum    = ws + 256 * 1024;            // 32*1024
  float* Sv      = ws + 288 * 1024;            // 32*1024
  float* attn    = ws + 320 * 1024;            // 32*1024
  float* Y       = ws + 352 * 1024;            // 32*1024

  // softmax over {256 x SCALE, 3840 x 0} collapses to two constants
  const float s  = 0.125f;  // HEAD_DIM^-0.5
  const float es = expf(s);
  const float Z  = 256.f * es + 3840.f;
  const float c1 = (es - 1.f) / Z;
  const float c2 = 1.f / Z;

  k_blocksum_partial<<<NBLK_TOT * RSPLIT, 256, 0, stream>>>(x, partial);
  k_blocksum_combine<<<32, 256, 0, stream>>>(partial, xsum);
  // Sv = xsum @ W_qkv[:,2048:3072] + 256*b_qkv[2048:3072]
  k_gemm_wmma<<<16, 256, 0, stream>>>(xsum, W_qkv, 3 * D_MODEL, 2 * D_MODEL,
                                      b_qkv, 2 * D_MODEL, 256.f, Sv);
  k_attn<<<8, 256, 0, stream>>>(Sv, attn, c1, c2);
  // Y = attn @ W_out + b_out
  k_gemm_wmma<<<16, 256, 0, stream>>>(attn, W_out, D_MODEL, 0, b_out, 0, 1.f, Y);
  k_broadcast<<<8192, 256, 0, stream>>>(Y, out);
}
